// TransformerModel_15762529976834
// MI455X (gfx1250) — matrix operand order repaired and checked by reading, not compiled, not run on hardware
//
#include <hip/hip_runtime.h>
#include <hip/hip_bf16.h>
#include <math.h>

// ---------------------------------------------------------------------------
// Model dims
// ---------------------------------------------------------------------------
#define LNUM 4
#define HNUM 8
#define DIM 1024
#define DFF 4096
#define RGB 1024
#define AUD 128
#define DIN 1152           // RGB + AUD
#define D_ATT 128
#define D_HOP 4
#define NCLS 3862
#define BB 32
#define TT 300
#define NTOK (BB * TT)     // 9600
#define DK (DIM / HNUM)    // 128

typedef __attribute__((ext_vector_type(16))) __bf16 v16bf;
typedef __attribute__((ext_vector_type(8)))  __bf16 v8bf;
typedef __attribute__((ext_vector_type(2)))  __bf16 v2bf;
typedef __attribute__((ext_vector_type(8)))  float  v8f;

// ---------------------------------------------------------------------------
// Generic batched GEMM:  C[z] = epilogue( A[z](MxK) * B[z](KxN) )
//   A f32 row-major (lda), B f32 (ldb; opB=1 reads B transposed: B[n*ldb+k])
//   epilogue: v = (acc + bias[n]) * scale ; act(0 none,1 relu,2 tanh); + Res
//   batch offsets: off = (z/zdiv)*s?1 + (z%zdiv)*s?2
// Tiles: 256 thr = 8 wave32, BM=64 BN=128 BK=32, wave tile 32x32,
//        4 x v_wmma_f32_16x16x32_bf16 per K-step per wave.
// LDS: double-buffered; A row-major [BM][BK], B transposed [BN][BK] so every
//      lane's fragment is contiguous -> pure ds_load_b128 gathers.
//      Single barrier per K-step: stage(k+1) overlaps compute(k).
// ---------------------------------------------------------------------------
#define BM 64
#define BN 128
#define BK 32
#define KPAD 8   // pitch 40 bf16 = 80B (16B-aligned rows)

__global__ __launch_bounds__(256) void gemm_bf16_wmma(
    const float* __restrict__ A, const float* __restrict__ Bw,
    const float* __restrict__ bias, const float* __restrict__ Res,
    float* __restrict__ C,
    int M, int N, int K, int lda, int ldb, int ldc,
    long long sA1, long long sA2, long long sB1, long long sB2,
    long long sC1, long long sC2, int zdiv,
    float scale, int act, int opB)
{
    __shared__ __bf16 As[2][BM][BK + KPAD];
    __shared__ __bf16 Bs[2][BN][BK + KPAD];   // transposed: [n][k]

    const int tid    = threadIdx.x;
    const int lane   = tid & 31;
    const int wid    = tid >> 5;
    const int half   = lane >> 4;       // 0 | 1
    const int l16    = lane & 15;
    const int waveM  = (wid >> 2) * 32; // {0,32}
    const int waveN  = (wid & 3) * 32;  // {0,32,64,96}
    const int blockM = blockIdx.y * BM;
    const int blockN = blockIdx.x * BN;
    const int z      = blockIdx.z;

    const long long aoff = (long long)(z / zdiv) * sA1 + (long long)(z % zdiv) * sA2;
    const long long boff = (long long)(z / zdiv) * sB1 + (long long)(z % zdiv) * sB2;
    const long long coff = (long long)(z / zdiv) * sC1 + (long long)(z % zdiv) * sC2;

    const bool fullMN = (blockM + BM <= M) && (blockN + BN <= N);

    // stage one BKxBM / BKxBN tile pair into LDS buffer `sel`
    auto stage = [&](int kt, int sel) {
        __bf16 (*Asb)[BK + KPAD] = As[sel];
        __bf16 (*Bsb)[BK + KPAD] = Bs[sel];
        if (fullMN && (kt + BK <= K)) {
            // -------- fast path: unconditional vector loads --------
#pragma unroll
            for (int it = 0; it < 4; ++it) {
                int pidx = tid + it * 256;
                int m   = pidx >> 4;          // 16 float2 per row
                int kk  = (pidx & 15) * 2;
                float2 v = *(const float2*)&A[aoff + (long long)(blockM + m) * lda + kt + kk];
                v2bf p; p[0] = (__bf16)v.x; p[1] = (__bf16)v.y;
                *(v2bf*)&Asb[m][kk] = p;
            }
            if (opB) {
#pragma unroll
                for (int it = 0; it < 8; ++it) {
                    int pidx = tid + it * 256;
                    int n  = pidx >> 4;
                    int kk = (pidx & 15) * 2;
                    float2 v = *(const float2*)&Bw[boff + (long long)(blockN + n) * ldb + kt + kk];
                    v2bf p; p[0] = (__bf16)v.x; p[1] = (__bf16)v.y;
                    *(v2bf*)&Bsb[n][kk] = p;
                }
            } else {
#pragma unroll
                for (int it = 0; it < 8; ++it) {
                    int pidx = tid + it * 256;
                    int kk = pidx >> 6;           // 64 float2 per k-row
                    int n  = (pidx & 63) * 2;
                    float2 v = *(const float2*)&Bw[boff + (long long)(kt + kk) * ldb + blockN + n];
                    Bsb[n][kk]     = (__bf16)v.x;
                    Bsb[n + 1][kk] = (__bf16)v.y;
                }
            }
        } else {
            // -------- slow path (edge tiles): clamped loads + select --------
#pragma unroll 1
            for (int it = 0; it < 8; ++it) {
                int idx = tid + it * 256;
                int m  = idx >> 5;
                int kk = idx & 31;
                int gm = blockM + m, gk = kt + kk;
                int cm = gm < M - 1 ? gm : M - 1;
                int ck = gk < K - 1 ? gk : K - 1;
                float v = A[aoff + (long long)cm * lda + ck];
                v = (gm < M && gk < K) ? v : 0.0f;
                Asb[m][kk] = (__bf16)v;
            }
#pragma unroll 1
            for (int it = 0; it < 16; ++it) {
                int idx = tid + it * 256;
                int kk = idx >> 7;
                int n  = idx & 127;
                int gk = kt + kk, gn = blockN + n;
                int cn = gn < N - 1 ? gn : N - 1;
                int ck = gk < K - 1 ? gk : K - 1;
                float v = opB ? Bw[boff + (long long)cn * ldb + ck]
                              : Bw[boff + (long long)ck * ldb + cn];
                v = (gk < K && gn < N) ? v : 0.0f;
                Bsb[n][kk] = (__bf16)v;
            }
        }
    };

    v8f acc00 = {}, acc01 = {}, acc10 = {}, acc11 = {};

    const int nk = (K + BK - 1) / BK;
    stage(0, 0);
    __syncthreads();

    for (int ik = 0; ik < nk; ++ik) {
        const int sel = ik & 1;
        if (ik + 1 < nk) stage((ik + 1) * BK, sel ^ 1);   // overlap with compute

        // ---- fragment gathers: contiguous 16B LDS chunks ----
        const int arow0 = waveM + l16;
        const int arow1 = waveM + 16 + l16;
        v8bf a0lo = *(const v8bf*)&As[sel][arow0][half * 8];
        v8bf a0hi = *(const v8bf*)&As[sel][arow0][16 + half * 8];
        v8bf a1lo = *(const v8bf*)&As[sel][arow1][half * 8];
        v8bf a1hi = *(const v8bf*)&As[sel][arow1][16 + half * 8];
        v16bf a0 = __builtin_shufflevector(a0lo, a0hi, 0,1,2,3,4,5,6,7,8,9,10,11,12,13,14,15);
        v16bf a1 = __builtin_shufflevector(a1lo, a1hi, 0,1,2,3,4,5,6,7,8,9,10,11,12,13,14,15);
        const int brow0 = waveN + l16;
        const int brow1 = waveN + 16 + l16;
        v8bf b0lo = *(const v8bf*)&Bs[sel][brow0][half * 8];
        v8bf b0hi = *(const v8bf*)&Bs[sel][brow0][16 + half * 8];
        v8bf b1lo = *(const v8bf*)&Bs[sel][brow1][half * 8];
        v8bf b1hi = *(const v8bf*)&Bs[sel][brow1][16 + half * 8];
        v16bf b0 = __builtin_shufflevector(b0lo, b0hi, 0,1,2,3,4,5,6,7,8,9,10,11,12,13,14,15);
        v16bf b1 = __builtin_shufflevector(b1lo, b1hi, 0,1,2,3,4,5,6,7,8,9,10,11,12,13,14,15);

        acc00 = __builtin_amdgcn_wmma_f32_16x16x32_bf16(false, a0, false, b0, (short)0, acc00, false, false);
        acc01 = __builtin_amdgcn_wmma_f32_16x16x32_bf16(false, a0, false, b1, (short)0, acc01, false, false);
        acc10 = __builtin_amdgcn_wmma_f32_16x16x32_bf16(false, a1, false, b0, (short)0, acc10, false, false);
        acc11 = __builtin_amdgcn_wmma_f32_16x16x32_bf16(false, a1, false, b1, (short)0, acc11, false, false);

        __syncthreads();   // single barrier per K-step
    }

    // ---- epilogue: C/D layout -> VGPR r holds M = half*8 + r, N = lane&15 ----
#pragma unroll
    for (int ti = 0; ti < 4; ++ti) {
        const v8f& acc = (ti == 0) ? acc00 : (ti == 1) ? acc01 : (ti == 2) ? acc10 : acc11;
        int mt = (ti >> 1) * 16, nt = (ti & 1) * 16;
        int gn = blockN + waveN + nt + l16;
#pragma unroll
        for (int r = 0; r < 8; ++r) {
            int gm = blockM + waveM + mt + half * 8 + r;
            if (gm < M && gn < N) {
                float v = acc[r];
                if (bias) v += bias[gn];
                v *= scale;
                if (act == 1) v = fmaxf(v, 0.0f);
                else if (act == 2) v = tanhf(v);
                long long ci = coff + (long long)gm * ldc + gn;
                if (Res) v += Res[ci];
                C[ci] = v;
            }
        }
    }
}

// ---------------------------------------------------------------------------
// feats = concat(rgb, aud) / 255      [NTOK, DIN]
// ---------------------------------------------------------------------------
__global__ void build_feats_k(const float* __restrict__ rgb,
                              const float* __restrict__ aud,
                              float* __restrict__ feats)
{
    long long idx = (long long)blockIdx.x * blockDim.x + threadIdx.x;
    long long total = (long long)NTOK * DIN;
    if (idx >= total) return;
    int c = (int)(idx % DIN);
    long long row = idx / DIN;
    float v = (c < RGB) ? rgb[row * RGB + c] : aud[row * AUD + (c - RGB)];
    feats[idx] = v * (1.0f / 255.0f);
}

// ---------------------------------------------------------------------------
// x[b,t,:] += PE(t,:)   (pairs: sin, cos interleaved)
// ---------------------------------------------------------------------------
__global__ void add_pe_k(float* __restrict__ x)
{
    long long idx = (long long)blockIdx.x * blockDim.x + threadIdx.x;
    long long total = (long long)NTOK * (DIM / 2);
    if (idx >= total) return;
    int pair = (int)(idx % (DIM / 2));
    long long bt = idx / (DIM / 2);
    int t = (int)(bt % TT);
    float div = __expf(-(float)(2 * pair) * (9.210340371976184f / (float)DIM));
    float ang = (float)t * div;
    float* p = x + bt * DIM + 2 * pair;
    p[0] += sinf(ang);
    p[1] += cosf(ang);
}

// ---------------------------------------------------------------------------
// LayerNorm (torch-style: unbiased std, a*(x-mean)/(std+eps)+b), D=1024
// ---------------------------------------------------------------------------
__global__ __launch_bounds__(256) void layernorm_k(
    const float* __restrict__ in, float* __restrict__ out,
    const float* __restrict__ g, const float* __restrict__ bta)
{
    int row = blockIdx.x;
    int tid = threadIdx.x;
    const float* p = in + (long long)row * DIM;
    float v[4];
    float s = 0.0f, q = 0.0f;
#pragma unroll
    for (int j = 0; j < 4; ++j) {
        v[j] = p[tid + j * 256];
        s += v[j];
        q += v[j] * v[j];
    }
#pragma unroll
    for (int off = 16; off; off >>= 1) {
        s += __shfl_xor(s, off, 32);
        q += __shfl_xor(q, off, 32);
    }
    __shared__ float ss[8], sq[8];
    __shared__ float smean, sinv;
    int wid = tid >> 5, lane = tid & 31;
    if (lane == 0) { ss[wid] = s; sq[wid] = q; }
    __syncthreads();
    if (tid == 0) {
        float S = 0.0f, Q = 0.0f;
        for (int w = 0; w < 8; ++w) { S += ss[w]; Q += sq[w]; }
        float mean = S * (1.0f / (float)DIM);
        float var  = (Q - (float)DIM * mean * mean) * (1.0f / (float)(DIM - 1));
        var = fmaxf(var, 0.0f);
        smean = mean;
        sinv  = 1.0f / (sqrtf(var) + 1e-6f);
    }
    __syncthreads();
    float mean = smean, inv = sinv;
    float* o = out + (long long)row * DIM;
#pragma unroll
    for (int j = 0; j < 4; ++j) {
        int d = tid + j * 256;
        o[d] = g[d] * (v[j] - mean) * inv + bta[d];
    }
}

// ---------------------------------------------------------------------------
// In-place softmax over contiguous rows; one wave32 per row
// ---------------------------------------------------------------------------
__global__ __launch_bounds__(256) void softmax_rows_k(float* __restrict__ S,
                                                      int nrows, int ncols)
{
    int row = blockIdx.x * (blockDim.x >> 5) + (threadIdx.x >> 5);
    if (row >= nrows) return;
    int lane = threadIdx.x & 31;
    float* p = S + (long long)row * ncols;
    float mx = -3.4e38f;
    for (int t = lane; t < ncols; t += 32) mx = fmaxf(mx, p[t]);
#pragma unroll
    for (int off = 16; off; off >>= 1) mx = fmaxf(mx, __shfl_xor(mx, off, 32));
    float sum = 0.0f;
    for (int t = lane; t < ncols; t += 32) {
        float e = __expf(p[t] - mx);
        p[t] = e;
        sum += e;
    }
#pragma unroll
    for (int off = 16; off; off >>= 1) sum += __shfl_xor(sum, off, 32);
    float inv = 1.0f / sum;
    for (int t = lane; t < ncols; t += 32) p[t] *= inv;
}

// ---------------------------------------------------------------------------
// alpha[b,hop,t] = softmax_t( logits[(b*T+t)*D_HOP + hop] );  128 rows
// ---------------------------------------------------------------------------
__global__ __launch_bounds__(256) void softmax_alpha_k(
    const float* __restrict__ logits, float* __restrict__ alpha)
{
    int row = blockIdx.x * (blockDim.x >> 5) + (threadIdx.x >> 5);
    if (row >= BB * D_HOP) return;
    int b = row >> 2, hop = row & 3;
    int lane = threadIdx.x & 31;
    const float* src = logits + (long long)b * TT * D_HOP + hop;
    float* dst = alpha + (long long)row * TT;
    float mx = -3.4e38f;
    for (int t = lane; t < TT; t += 32) mx = fmaxf(mx, src[(long long)t * D_HOP]);
#pragma unroll
    for (int off = 16; off; off >>= 1) mx = fmaxf(mx, __shfl_xor(mx, off, 32));
    float sum = 0.0f;
    for (int t = lane; t < TT; t += 32) {
        float e = __expf(src[(long long)t * D_HOP] - mx);
        dst[t] = e;
        sum += e;
    }
#pragma unroll
    for (int off = 16; off; off >>= 1) sum += __shfl_xor(sum, off, 32);
    float inv = 1.0f / sum;
    for (int t = lane; t < TT; t += 32) dst[t] *= inv;
}

// ---------------------------------------------------------------------------
// Host-side launch
// ---------------------------------------------------------------------------
static inline void gemm(hipStream_t st,
                        const float* A, const float* Bw, const float* bias,
                        const float* Res, float* C,
                        int M, int N, int K, int lda, int ldb, int ldc,
                        long long sA1, long long sA2, long long sB1, long long sB2,
                        long long sC1, long long sC2, int batch, int zdiv,
                        float scale, int act, int opB)
{
    dim3 grid((N + BN - 1) / BN, (M + BM - 1) / BM, batch);
    gemm_bf16_wmma<<<grid, 256, 0, st>>>(A, Bw, bias, Res, C,
                                         M, N, K, lda, ldb, ldc,
                                         sA1, sA2, sB1, sB2, sC1, sC2, zdiv,
                                         scale, act, opB);
}

extern "C" void kernel_launch(void* const* d_in, const int* in_sizes, int n_in,
                              void* d_out, int out_size, void* d_ws, size_t ws_size,
                              hipStream_t stream)
{
    (void)in_sizes; (void)n_in; (void)out_size; (void)ws_size;

    const float* rgb    = (const float*)d_in[0];
    const float* aud    = (const float*)d_in[1];
    const float* emW    = (const float*)d_in[2];
    const float* emB    = (const float*)d_in[3];
    const float* Wq     = (const float*)d_in[4];
    const float* bq     = (const float*)d_in[5];
    const float* Wk     = (const float*)d_in[6];
    const float* bk     = (const float*)d_in[7];
    const float* Wv     = (const float*)d_in[8];
    const float* bv     = (const float*)d_in[9];
    const float* Wo     = (const float*)d_in[10];
    const float* bo     = (const float*)d_in[11];
    const float* ln1a   = (const float*)d_in[12];
    const float* ln1b   = (const float*)d_in[13];
    const float* W1f    = (const float*)d_in[14];
    const float* b1f    = (const float*)d_in[15];
    const float* W2f    = (const float*)d_in[16];
    const float* b2f    = (const float*)d_in[17];
    const float* ln2a   = (const float*)d_in[18];
    const float* ln2b   = (const float*)d_in[19];
    const float* fna    = (const float*)d_in[20];
    const float* fnb    = (const float*)d_in[21];
    const float* cw1    = (const float*)d_in[22];
    const float* cb1    = (const float*)d_in[23];
    const float* cw2    = (const float*)d_in[24];
    const float* cb2    = (const float*)d_in[25];
    const float* cw3    = (const float*)d_in[26];
    const float* cb3    = (const float*)d_in[27];
    const float* cw4    = (const float*)d_in[28];
    const float* cb4    = (const float*)d_in[29];

    // workspace layout (floats)
    const long long NX = (long long)NTOK * DIM;           // 9,830,400
    float* wsf  = (float*)d_ws;
    float* X    = wsf + 0 * NX;
    float* X2   = wsf + 1 * NX;
    float* XN   = wsf + 2 * NX;
    float* Q    = wsf + 3 * NX;
    float* Kb   = wsf + 4 * NX;
    float* V    = wsf + 5 * NX;
    float* O    = wsf + 6 * NX;
    float* S    = wsf + 7 * NX;                           // [B,H,T,T]
    float* HFF  = S + (long long)BB * HNUM * TT * TT;     // [NTOK,DFF]
    float* FEATS = HFF;                                   // alias (pre-layers only)
    float* T1   = Q;                                      // alias, post-layers
    float* ALOG = Kb;                                     // alias, post-layers
    float* POOL = V;                                      // alias, post-layers
    float* CT   = O;                                      // alias, post-layers

    float* out_main  = (float*)d_out;                     // [B, NCLS]
    float* alpha_out = out_main + (long long)BB * NCLS;   // [B, D_HOP, T]

    const float inv_sqrt_dk = 0.088388347648318447f;      // 1/sqrt(128)
    const float sqrt_d = 32.0f;                           // sqrt(1024)

    // ---- embedding ----
    {
        long long tot = (long long)NTOK * DIN;
        build_feats_k<<<(unsigned)((tot + 255) / 256), 256, 0, stream>>>(rgb, aud, FEATS);
        gemm(stream, FEATS, emW, emB, nullptr, X,
             NTOK, DIM, DIN, DIN, DIM, DIM,
             0, 0, 0, 0, 0, 0, 1, 1, sqrt_d, 0, 0);
        long long tpe = (long long)NTOK * (DIM / 2);
        add_pe_k<<<(unsigned)((tpe + 255) / 256), 256, 0, stream>>>(X);
    }

    // ---- transformer layers ----
    for (int i = 0; i < LNUM; ++i) {
        const float* wq = Wq + (long long)i * DIM * DIM;
        const float* wk = Wk + (long long)i * DIM * DIM;
        const float* wv = Wv + (long long)i * DIM * DIM;
        const float* wo = Wo + (long long)i * DIM * DIM;
        const float* w1 = W1f + (long long)i * DIM * DFF;
        const float* w2 = W2f + (long long)i * DFF * DIM;

        layernorm_k<<<NTOK, 256, 0, stream>>>(X, XN, ln1a + i * DIM, ln1b + i * DIM);

        gemm(stream, XN, wq, bq + i * DIM, nullptr, Q,
             NTOK, DIM, DIM, DIM, DIM, DIM, 0, 0, 0, 0, 0, 0, 1, 1, 1.0f, 0, 0);
        gemm(stream, XN, wk, bk + i * DIM, nullptr, Kb,
             NTOK, DIM, DIM, DIM, DIM, DIM, 0, 0, 0, 0, 0, 0, 1, 1, 1.0f, 0, 0);
        gemm(stream, XN, wv, bv + i * DIM, nullptr, V,
             NTOK, DIM, DIM, DIM, DIM, DIM, 0, 0, 0, 0, 0, 0, 1, 1, 1.0f, 0, 0);

        // scores[b,h] = Q[b,:,h] @ K[b,:,h]^T / sqrt(dk)   (z = b*H + h)
        gemm(stream, Q, Kb, nullptr, nullptr, S,
             TT, TT, DK, DIM, DIM, TT,
             (long long)TT * DIM, DK,
             (long long)TT * DIM, DK,
             (long long)HNUM * TT * TT, (long long)TT * TT,
             BB * HNUM, HNUM, inv_sqrt_dk, 0, /*opB=*/1);

        softmax_rows_k<<<(BB * HNUM * TT + 7) / 8, 256, 0, stream>>>(S, BB * HNUM * TT, TT);

        // O[b,:,h] = P[b,h] @ V[b,:,h]
        gemm(stream, S, V, nullptr, nullptr, O,
             TT, DK, TT, TT, DIM, DIM,
             (long long)HNUM * TT * TT, (long long)TT * TT,
             (long long)TT * DIM, DK,
             (long long)TT * DIM, DK,
             BB * HNUM, HNUM, 1.0f, 0, 0);

        // X2 = X + O @ Wo + bo
        gemm(stream, O, wo, bo + i * DIM, X, X2,
             NTOK, DIM, DIM, DIM, DIM, DIM, 0, 0, 0, 0, 0, 0, 1, 1, 1.0f, 0, 0);

        layernorm_k<<<NTOK, 256, 0, stream>>>(X2, XN, ln2a + i * DIM, ln2b + i * DIM);

        // HFF = relu(XN @ W1 + b1)
        gemm(stream, XN, w1, b1f + i * DFF, nullptr, HFF,
             NTOK, DFF, DIM, DIM, DFF, DFF, 0, 0, 0, 0, 0, 0, 1, 1, 1.0f, 1, 0);
        // X = X2 + HFF @ W2 + b2
        gemm(stream, HFF, w2, b2f + i * DIM, X2, X,
             NTOK, DIM, DFF, DFF, DIM, DIM, 0, 0, 0, 0, 0, 0, 1, 1, 1.0f, 0, 0);
    }

    // ---- final norm + attention-pooling classifier ----
    layernorm_k<<<NTOK, 256, 0, stream>>>(X, XN, fna, fnb);

    // T1 = tanh(XN @ cw1 + cb1)                 [NTOK, 128]
    gemm(stream, XN, cw1, cb1, nullptr, T1,
         NTOK, D_ATT, DIM, DIM, D_ATT, D_ATT, 0, 0, 0, 0, 0, 0, 1, 1, 1.0f, 2, 0);
    // ALOG = T1 @ cw2 + cb2                     [NTOK, 4]
    gemm(stream, T1, cw2, cb2, nullptr, ALOG,
         NTOK, D_HOP, D_ATT, D_ATT, D_HOP, D_HOP, 0, 0, 0, 0, 0, 0, 1, 1, 1.0f, 0, 0);
    // alpha = softmax over t  -> second output
    softmax_alpha_k<<<(BB * D_HOP + 7) / 8, 256, 0, stream>>>(ALOG, alpha_out);

    // POOL[b] = alpha[b] (4x300) @ XN[b] (300x1024)
    gemm(stream, alpha_out, XN, nullptr, nullptr, POOL,
         D_HOP, DIM, TT, TT, DIM, DIM,
         (long long)D_HOP * TT, 0,
         (long long)TT * DIM, 0,
         (long long)D_HOP * DIM, 0,
         BB, 1, 1.0f, 0, 0);

    // CT = tanh(POOL @ cw3 + cb3)               [B, DFF]
    gemm(stream, POOL, cw3, cb3, nullptr, CT,
         BB, DFF, D_HOP * DIM, D_HOP * DIM, DFF, DFF, 0, 0, 0, 0, 0, 0, 1, 1, 1.0f, 2, 0);
    // out = CT @ cw4 + cb4                      [B, NCLS]
    gemm(stream, CT, cw4, cb4, nullptr, out_main,
         BB, NCLS, DFF, DFF, NCLS, NCLS, 0, 0, 0, 0, 0, 0, 1, 1, 1.0f, 0, 0);
}
